// TemporalHeterogeneousAttention_82308753261274
// MI455X (gfx1250) — matrix operand order repaired and checked by reading, not compiled, not run on hardware
//
#include <hip/hip_runtime.h>
#include <hip/hip_bf16.h>
#include <math.h>

typedef __attribute__((ext_vector_type(16))) _Float16 v16h;
typedef __attribute__((ext_vector_type(8)))  float    v8f;

static constexpr int D = 256, H = 4, R = 4, TD = 64, NNODES = 50000;
static constexpr int ADIM = 2 * D + TD;           // 576
static constexpr float SLOPE = 0.2f;

// counter layout inside ws (ints): [0..3]=count [4..7]=cursor [8..11]=seg_start [12..16]=tile_off
static constexpr int CI_CNT = 0, CI_CUR = 4, CI_SEG = 8, CI_TOFF = 12;

__device__ __forceinline__ void atomicMaxF(float* addr, float val) {
    unsigned int* u = (unsigned int*)addr;
    unsigned int old = __hip_atomic_load(u, __ATOMIC_RELAXED, __HIP_MEMORY_SCOPE_AGENT);
    while (__uint_as_float(old) < val) {
        unsigned int assumed = old;
        old = atomicCAS(u, assumed, __float_as_uint(val));
        if (old == assumed) break;
    }
}

// ---------------- setup kernels ----------------

__global__ void k_init(float* __restrict__ out, float* __restrict__ maxd,
                       float* __restrict__ denom, int* __restrict__ ctr,
                       int tot_out, int tot_nh) {
    int i = blockIdx.x * blockDim.x + threadIdx.x;
    if (i < tot_out) out[i] = 0.0f;
    if (i < tot_nh) { maxd[i] = -3.0e38f; denom[i] = 0.0f; }
    if (i < 32) ctr[i] = 0;
}

__global__ void k_count(const int* __restrict__ rel, int* __restrict__ ctr, int E) {
    int i = blockIdx.x * blockDim.x + threadIdx.x;
    if (i < E) atomicAdd(&ctr[CI_CNT + rel[i]], 1);
}

__global__ void k_scan(int* __restrict__ ctr) {
    if (threadIdx.x == 0 && blockIdx.x == 0) {
        int s = 0, t = 0;
        for (int r = 0; r < R; ++r) {
            ctr[CI_SEG + r] = s; ctr[CI_CUR + r] = s; s += ctr[CI_CNT + r];
            ctr[CI_TOFF + r] = t; t += (ctr[CI_CNT + r] + 15) / 16;
        }
        ctr[CI_TOFF + R] = t;
    }
}

__global__ void k_scatter(const int* __restrict__ rel, int* __restrict__ ctr,
                          int* __restrict__ sorted_eid, int E) {
    int i = blockIdx.x * blockDim.x + threadIdx.x;
    if (i < E) {
        int p = atomicAdd(&ctr[CI_CUR + rel[i]], 1);
        sorted_eid[p] = i;
    }
}

__global__ void k_convW(const float* __restrict__ W, _Float16* __restrict__ W16, int n) {
    int i = blockIdx.x * blockDim.x + threadIdx.x;
    if (i < n) W16[i] = (_Float16)W[i];
}

// ---------------- main WMMA tile kernel ----------------
// One block = 16 edges of one relation. 8 waves; wave w computes Wv & Wu
// output tiles for N-columns [16w,16w+16) and [16(w+8),16(w+8)+16).
// Across the 8 waves all 256 N-columns are covered exactly once, so each
// block streams W16[r] (128 KB) from L2 exactly once per 16 edges.

__global__ __launch_bounds__(256) void k_tile(
    const float* __restrict__ hv, const float* __restrict__ hu,
    const float* __restrict__ dt, const long long* __restrict__ tgt,
    const float* __restrict__ bias, const float* __restrict__ a,
    const float* __restrict__ timew, const _Float16* __restrict__ W16,
    const int* __restrict__ ctr, const int* __restrict__ sorted_eid,
    _Float16* __restrict__ Wu16, float* __restrict__ ml,
    int* __restrict__ tgt_s, int* __restrict__ rel_s,
    float* __restrict__ maxd, int E)
{
    __shared__ _Float16 sh_hv[16][264];
    __shared__ _Float16 sh_hu[16][264];
    __shared__ float    sh_wv[16][264];
    __shared__ float    sh_wu[16][264];
    __shared__ float    sh_phi[16][TD];
    __shared__ int      sh_eid[16];
    __shared__ int      sh_tgt[16];
    __shared__ float    sh_dt[16];

    int b = blockIdx.x;
    int total = ctr[CI_TOFF + R];
    if (b >= total) return;                       // uniform exit, pre-barrier
    int r = 0;
    while (r < R - 1 && b >= ctr[CI_TOFF + r + 1]) ++r;
    int trel   = b - ctr[CI_TOFF + r];
    int estart = ctr[CI_SEG + r] + trel * 16;
    int cnt    = ctr[CI_CNT + r] - trel * 16; if (cnt > 16) cnt = 16;

    int tid = threadIdx.x;
    if (tid < 16) {
        int m  = tid;
        int mm = m < cnt ? m : (cnt - 1);         // pad by duplicating last edge
        int eid = sorted_eid[estart + mm];
        sh_eid[m] = eid;
        sh_tgt[m] = (int)tgt[eid];
        sh_dt[m]  = dt[eid];
    }
    __syncthreads();

    // gather 16 x 256 rows of h_v / h_u, convert fp32 -> fp16 into LDS
    {
        int m  = tid >> 4;
        int d0 = (tid & 15) * 16;
        const float* sv = hv + (size_t)sh_eid[m] * D + d0;
        const float* su = hu + (size_t)sh_eid[m] * D + d0;
        #pragma unroll
        for (int j = 0; j < 16; ++j) {
            sh_hv[m][d0 + j] = (_Float16)sv[j];
            sh_hu[m][d0 + j] = (_Float16)su[j];
        }
    }
    // time encoding phi(t) = cos(dt * w): 16x64 entries, 4 per thread
    #pragma unroll
    for (int i = 0; i < 4; ++i) {
        int idx = tid + i * 256;
        int m = idx >> 6, j = idx & 63;
        sh_phi[m][j] = cosf(sh_dt[m] * timew[j]);
    }
    __syncthreads();

    // ---- WMMA: [16 edges x 256] = A(16x256,f16) x W[r]^T(256x256,f16) ----
    int w    = tid >> 5;
    int lane = tid & 31;
    int mcol = lane & 15;
    int hi   = lane >> 4;                          // 0 or 1
    v8f av0 = {}, av1 = {}, au0 = {}, au1 = {};
    int n0 = w * 16 + mcol;
    int n1 = (w + 8) * 16 + mcol;
    const _Float16* w0 = W16 + ((size_t)r * D + n0) * D;
    const _Float16* w1 = W16 + ((size_t)r * D + n1) * D;

    for (int kc = 0; kc < 8; ++kc) {
        int kA = kc * 32 + hi * 8;                 // A layout: lane<16 K0-7/16-23, lane>=16 K8-15/24-31
        int kB = kc * 32 + hi * 8;                 // B layout: lane<16 K0-7/16-23, lane>=16 K8-15/24-31
        v16h Av, Au, B0, B1;
        #pragma unroll
        for (int j = 0; j < 8; ++j) {
            Av[j]     = sh_hv[mcol][kA + j];
            Av[j + 8] = sh_hv[mcol][kA + 16 + j];
            Au[j]     = sh_hu[mcol][kA + j];
            Au[j + 8] = sh_hu[mcol][kA + 16 + j];
        }
        #pragma unroll
        for (int j = 0; j < 8; ++j) { B0[j] = w0[kB + j]; B1[j] = w1[kB + j]; B0[j + 8] = w0[kB + 16 + j]; B1[j + 8] = w1[kB + 16 + j]; }
        av0 = __builtin_amdgcn_wmma_f32_16x16x32_f16(false, Av, false, B0, (short)0, av0, false, false);
        av1 = __builtin_amdgcn_wmma_f32_16x16x32_f16(false, Av, false, B1, (short)0, av1, false, false);
        au0 = __builtin_amdgcn_wmma_f32_16x16x32_f16(false, Au, false, B0, (short)0, au0, false, false);
        au1 = __builtin_amdgcn_wmma_f32_16x16x32_f16(false, Au, false, B1, (short)0, au1, false, false);
    }
    float bb0 = bias[r * D + n0], bb1 = bias[r * D + n1];
    #pragma unroll
    for (int j = 0; j < 8; ++j) {                  // C/D layout: vgpr j -> M = j + 8*hi, N fixed
        int mr = j + hi * 8;
        sh_wv[mr][n0] = av0[j] + bb0;
        sh_wv[mr][n1] = av1[j] + bb1;
        sh_wu[mr][n0] = au0[j] + bb0;
        sh_wu[mr][n1] = au1[j] + bb1;
    }
    __syncthreads();

    // write Wu (f16) to workspace at sorted position, coalesced
    {
        int m  = tid >> 4;
        int d0 = (tid & 15) * 16;
        if (m < cnt) {
            _Float16* dst = Wu16 + (size_t)(estart + m) * D + d0;
            #pragma unroll
            for (int j = 0; j < 16; ++j) dst[j] = (_Float16)sh_wu[m][d0 + j];
        }
    }
    if (tid < 16 && tid < cnt) {
        tgt_s[estart + tid] = sh_tgt[tid];
        rel_s[estart + tid] = r;
    }

    // attention logits: all 256 threads. Thread -> (edge m, head h, slice q);
    // 4 consecutive lanes split the 576-dim concat dot into 144-dim slices,
    // reduced with wave32 shuffles (ds_permute path), lane q==0 commits.
    {
        int m = tid >> 4;          // 0..15 edge
        int h = (tid >> 2) & 3;    // head
        int q = tid & 3;           // slice within (m,h)
        const float* ar = a + ((size_t)r * H + h) * ADIM;
        int lo = q * 144, hi2 = lo + 144;
        float s = 0.0f;
        int e1 = hi2 < D ? hi2 : D;                       // Wv segment [lo, min(hi,256))
        for (int g = lo; g < e1; ++g)  s = fmaf(sh_wv[m][g], ar[g], s);
        int b2 = lo > D ? lo : D;                         // Wu segment
        int e2 = hi2 < 2 * D ? hi2 : 2 * D;
        for (int g = b2; g < e2; ++g)  s = fmaf(sh_wu[m][g - D], ar[g], s);
        int b3 = lo > 2 * D ? lo : 2 * D;                 // phi segment
        for (int g = b3; g < hi2; ++g) s = fmaf(sh_phi[m][g - 2 * D], ar[g], s);
        s += __shfl_xor(s, 1, 32);
        s += __shfl_xor(s, 2, 32);
        if (q == 0 && m < cnt) {
            float lg = s > 0.0f ? s : SLOPE * s;
            ml[(size_t)(estart + m) * H + h] = lg;
            atomicMaxF(&maxd[((size_t)r * NNODES + sh_tgt[m]) * H + h], lg);
        }
    }
}

// ---------------- softmax denominator ----------------

__global__ void k_denom(const float* __restrict__ ml, const int* __restrict__ tgt_s,
                        const int* __restrict__ rel_s, const float* __restrict__ maxd,
                        float* __restrict__ denom, int EH) {
    int i = blockIdx.x * blockDim.x + threadIdx.x;
    if (i >= EH) return;
    int pos = i >> 2, h = i & 3;
    int r = rel_s[pos], t = tgt_s[pos];
    size_t idx = ((size_t)r * NNODES + t) * H + h;
    float e = expf(ml[i] - maxd[idx]);
    atomicAdd(&denom[idx], e);
}

// ---------------- weighted scatter to output ----------------
// block handles 8 edges x 256 dims; alpha computed once per (edge, head) in LDS

__global__ __launch_bounds__(256) void k_out(
    const _Float16* __restrict__ Wu16, const float* __restrict__ ml,
    const int* __restrict__ tgt_s, const int* __restrict__ rel_s,
    const float* __restrict__ maxd, const float* __restrict__ denom,
    float* __restrict__ out, int E)
{
    __shared__ float alph[8][H];
    __shared__ int   tg[8];
    int base = blockIdx.x * 8;
    int tid  = threadIdx.x;
    if (tid < 32) {
        int i = tid >> 2, h = tid & 3;
        int pos = base + i;
        float av = 0.0f; int t = 0;
        if (pos < E) {
            int r = rel_s[pos]; t = tgt_s[pos];
            size_t idx = ((size_t)r * NNODES + t) * H + h;
            av = expf(ml[(size_t)pos * H + h] - maxd[idx]) / (denom[idx] + 1e-9f);
        }
        alph[i][h] = av;
        if (h == 0) tg[i] = t;
    }
    __syncthreads();
    int d = tid, hh = d >> 6;
    for (int i = 0; i < 8; ++i) {
        int pos = base + i;
        if (pos >= E) break;
        float v = (float)Wu16[(size_t)pos * D + d] * alph[i][hh];
        atomicAdd(&out[(size_t)tg[i] * D + d], v);
    }
}

// ---------------- host launcher ----------------

extern "C" void kernel_launch(void* const* d_in, const int* in_sizes, int n_in,
                              void* d_out, int out_size, void* d_ws, size_t ws_size,
                              hipStream_t stream) {
    (void)n_in; (void)out_size; (void)ws_size;
    const float*     hv    = (const float*)d_in[0];
    const float*     hu    = (const float*)d_in[1];
    const int*       rel   = (const int*)d_in[2];
    const float*     dt    = (const float*)d_in[3];
    const long long* tgt   = (const long long*)d_in[4];
    const float*     W     = (const float*)d_in[6];
    const float*     bias  = (const float*)d_in[7];
    const float*     a     = (const float*)d_in[8];
    const float*     timew = (const float*)d_in[9];
    float*           out   = (float*)d_out;
    const int E = in_sizes[2];

    // workspace carve-up (256B aligned chunks)
    char* basep = (char*)d_ws;
    size_t off = 0;
    auto take = [&](size_t bytes) -> char* {
        char* p = basep + off;
        off = (off + bytes + 255) & ~(size_t)255;
        return p;
    };
    int*      ctr        = (int*)take(32 * sizeof(int));
    int*      sorted_eid = (int*)take((size_t)E * 4);
    int*      tgt_s      = (int*)take((size_t)E * 4);
    int*      rel_s      = (int*)take((size_t)E * 4);
    float*    ml         = (float*)take((size_t)E * H * 4);
    float*    maxd       = (float*)take((size_t)R * NNODES * H * 4);
    float*    denom      = (float*)take((size_t)R * NNODES * H * 4);
    _Float16* W16        = (_Float16*)take((size_t)R * D * D * 2);
    _Float16* Wu16       = (_Float16*)take((size_t)E * D * 2);

    const int tot_out = NNODES * D;
    const int tot_nh  = R * NNODES * H;

    k_init   <<<(tot_out + 255) / 256, 256, 0, stream>>>(out, maxd, denom, ctr, tot_out, tot_nh);
    k_count  <<<(E + 255) / 256,       256, 0, stream>>>(rel, ctr, E);
    k_convW  <<<(R * D * D + 255) / 256, 256, 0, stream>>>(W, W16, R * D * D);
    k_scan   <<<1, 1, 0, stream>>>(ctr);
    k_scatter<<<(E + 255) / 256,       256, 0, stream>>>(rel, ctr, sorted_eid, E);

    const int tiles_max = (E + 15) / 16 + R;      // fixed upper bound; blocks self-map & exit
    k_tile   <<<tiles_max, 256, 0, stream>>>(hv, hu, dt, tgt, bias, a, timew, W16, ctr,
                                             sorted_eid, Wu16, ml, tgt_s, rel_s, maxd, E);
    k_denom  <<<(E * H + 255) / 256,   256, 0, stream>>>(ml, tgt_s, rel_s, maxd, denom, E * H);
    k_out    <<<(E + 7) / 8,           256, 0, stream>>>(Wu16, ml, tgt_s, rel_s, maxd, denom, out, E);
}
